// DistributedMonitoredAttention_51616916964075
// MI455X (gfx1250) — matrix operand order repaired and checked by reading, not compiled, not run on hardware
//
#include <hip/hip_runtime.h>
#include <hip/hip_bf16.h>
#include <math.h>
#include <stdint.h>

// MI455X / gfx1250, wave32. bf16 WMMA (16x16x32) with f32 accumulation.

typedef __bf16 bf16_t;
typedef bf16_t v16bf __attribute__((ext_vector_type(16)));
typedef bf16_t v8bf  __attribute__((ext_vector_type(8)));
typedef bf16_t v4bf  __attribute__((ext_vector_type(4)));
typedef float  v8f   __attribute__((ext_vector_type(8)));
typedef int    v4i   __attribute__((ext_vector_type(4)));

typedef v4i __attribute__((address_space(1)))* gv4i_p;   // global
typedef v4i __attribute__((address_space(3)))* lv4i_p;   // LDS

#define D_MODEL 2048
#define SEQ     2048
#define BATCH   4
#define HEADS   16
#define DK      128
#define MTOT    (BATCH * SEQ)   // 8192 flattened rows

#ifndef __has_builtin
#define __has_builtin(x) 0
#endif
#if __has_builtin(__builtin_amdgcn_global_load_async_to_lds_b128) && \
    __has_builtin(__builtin_amdgcn_s_wait_asynccnt)
#define HAVE_ASYNC_LDS 1
#else
#define HAVE_ASYNC_LDS 0
#endif

union FragU { v16bf v; v8bf h[2]; };

// A-matrix fragment (16x32, MxK): lane r=lane&15 holds row r; lanes<16 take
// K = k0+0..7 & k0+16..23, lanes>=16 take K = k0+8..15 & k0+24..31.
__device__ __forceinline__ v16bf load_fragA(const bf16_t* base, int row0, int ld, int k0) {
    const int lane = threadIdx.x & 31;
    const int r = lane & 15, sel = lane >> 4;
    const bf16_t* p = base + (size_t)(row0 + r) * ld + (k0 + sel * 8);
    FragU f;
    f.h[0] = *(const v8bf*)(p);
    f.h[1] = *(const v8bf*)(p + 16);
    return f.v;
}

// B-matrix fragment (32x16, KxN): lane holds column n=lane&15; lanes<16 hold
// K = k0+0..7 & k0+16..23, lanes>=16 hold K = k0+8..15 & k0+24..31 (along a row of the
// row-major source, since column n of B == row n of W / K / V^T).
__device__ __forceinline__ v16bf load_fragB(const bf16_t* base, int n0, int ld, int k0) {
    const int lane = threadIdx.x & 31;
    const int n = lane & 15, sel = lane >> 4;
    const bf16_t* p = base + (size_t)(n0 + n) * ld + (k0 + sel * 8);
    FragU f;
    f.h[0] = *(const v8bf*)(p);
    f.h[1] = *(const v8bf*)(p + 16);
    return f.v;
}

__device__ __forceinline__ v8f wmma_bf16(v16bf a, v16bf b, v8f c) {
    return __builtin_amdgcn_wmma_f32_16x16x32_bf16(false, a, false, b, (short)0, c, false, false);
}

// 16-byte chunk copy into LDS: ASYNCcnt-tracked DMA on gfx1250, sync fallback.
__device__ __forceinline__ void copy_b128_to_lds(const bf16_t* gp, bf16_t* lp) {
#if HAVE_ASYNC_LDS
    __builtin_amdgcn_global_load_async_to_lds_b128(
        reinterpret_cast<gv4i_p>(reinterpret_cast<uintptr_t>(gp)),
        reinterpret_cast<lv4i_p>(reinterpret_cast<uintptr_t>(lp)),
        0, 0);
#else
    *(v8bf*)lp = *(const v8bf*)gp;
#endif
}

// Stage a 128x32 bf16 tile (row-major, ld=gld) into LDS [128][32].
// 128 threads; each thread owns one row (4 x b128 chunks).
__device__ __forceinline__ void stage_tile(const bf16_t* __restrict__ g, int gld,
                                           int row0, int k0, bf16_t* l) {
    const int row = threadIdx.x;  // 0..127
    const bf16_t* gp = g + (size_t)(row0 + row) * gld + k0;
    bf16_t* lp = l + row * 32;
    #pragma unroll
    for (int t = 0; t < 4; ++t)
        copy_b128_to_lds(gp + t * 8, lp + t * 8);
}

// ---------------------------------------------------------------- converts
__global__ __launch_bounds__(256) void cvt_f32_bf16(const float* __restrict__ in,
                                                    bf16_t* __restrict__ out, size_t n) {
    size_t i = ((size_t)blockIdx.x * 256 + threadIdx.x) * 4;
    if (i + 3 < n) {
        float4 f = *(const float4*)(in + i);
        v4bf o;
        o[0] = (bf16_t)f.x; o[1] = (bf16_t)f.y; o[2] = (bf16_t)f.z; o[3] = (bf16_t)f.w;
        *(v4bf*)(out + i) = o;
    }
}

// ---------------------------------------------------------------- GEMM
// C[M,N] = A[M,K] * W[N,K]^T, M=8192, N=K=2048.
// Block: 128 thr = 4 waves as 2(M) x 2(N); block tile 128x128; wave tile 64x64.
// Double-buffered async LDS staging of the 128x32 A and B K-slabs.
// MODE 0: Q  -> bf16 [B,H,S,Dk], scaled by 1/sqrt(Dk)
// MODE 1: K  -> bf16 [B,H,S,Dk]
// MODE 2: Vt -> bf16 [B,H,Dk,S] (transposed store)
// MODE 3: f32 row-major [M,N] (final projection to d_out)
template<int MODE>
__global__ __launch_bounds__(128) void gemm_bf16(const bf16_t* __restrict__ A,
                                                 const bf16_t* __restrict__ W,
                                                 void* __restrict__ outp) {
    __shared__ bf16_t As[2][128 * 32];
    __shared__ bf16_t Bs[2][128 * 32];

    const int waveId = threadIdx.x >> 5;
    const int lane   = threadIdx.x & 31;
    const int blockM = blockIdx.x * 128;
    const int blockN = blockIdx.y * 128;
    const int wm     = (waveId >> 1) * 64;   // wave origin inside block tile
    const int wn     = (waveId & 1) * 64;

    v8f zero = {};
    v8f acc[4][4];
    #pragma unroll
    for (int i = 0; i < 4; ++i)
        #pragma unroll
        for (int j = 0; j < 4; ++j) acc[i][j] = zero;

    stage_tile(A, D_MODEL, blockM, 0, As[0]);
    stage_tile(W, D_MODEL, blockN, 0, Bs[0]);

    int buf = 0;
    for (int k0 = 0; k0 < D_MODEL; k0 += 32, buf ^= 1) {
        if (k0 + 32 < D_MODEL) {
            stage_tile(A, D_MODEL, blockM, k0 + 32, As[buf ^ 1]);
            stage_tile(W, D_MODEL, blockN, k0 + 32, Bs[buf ^ 1]);
#if HAVE_ASYNC_LDS
            __builtin_amdgcn_s_wait_asynccnt(8);  // prior slab's 8 asyncs done (in-order)
#endif
        } else {
#if HAVE_ASYNC_LDS
            __builtin_amdgcn_s_wait_asynccnt(0);
#endif
        }
        __syncthreads();   // current buffer visible to all waves

        v16bf a[4];
        #pragma unroll
        for (int i = 0; i < 4; ++i) a[i] = load_fragA(As[buf], wm + 16 * i, 32, 0);
        #pragma unroll
        for (int j = 0; j < 4; ++j) {
            v16bf b = load_fragB(Bs[buf], wn + 16 * j, 32, 0);
            #pragma unroll
            for (int i = 0; i < 4; ++i) acc[i][j] = wmma_bf16(a[i], b, acc[i][j]);
        }
        __syncthreads();   // everyone done reading before buffer is overwritten
    }

    const int sel = lane >> 4, nl = lane & 15;
    #pragma unroll
    for (int i = 0; i < 4; ++i) {
        #pragma unroll
        for (int j = 0; j < 4; ++j) {
            #pragma unroll
            for (int r = 0; r < 8; ++r) {
                const int m = blockM + wm + i * 16 + r + sel * 8;
                const int n = blockN + wn + j * 16 + nl;
                float v = acc[i][j][r];
                if (MODE == 3) {
                    ((float*)outp)[(size_t)m * D_MODEL + n] = v;
                } else {
                    const int bb = m >> 11, s = m & (SEQ - 1);
                    const int h  = n >> 7,  dk = n & (DK - 1);
                    bf16_t* o = (bf16_t*)outp;
                    if (MODE == 0) {
                        v *= 0.08838834764831845f; // 1/sqrt(128)
                        o[(((size_t)(bb * HEADS + h)) * SEQ + s) * DK + dk] = (bf16_t)v;
                    } else if (MODE == 1) {
                        o[(((size_t)(bb * HEADS + h)) * SEQ + s) * DK + dk] = (bf16_t)v;
                    } else { // MODE 2: transposed V
                        o[(((size_t)(bb * HEADS + h)) * DK + dk) * SEQ + s] = (bf16_t)v;
                    }
                }
            }
        }
    }
}

// ---------------------------------------------------------------- flash attention
// Grid: (SEQ/64, B*H); block 128 thr = 4 waves; each wave owns 16 query rows.
// Q pre-scaled by 1/sqrt(Dk). Online softmax in f32; P via per-wave LDS tile.
__global__ __launch_bounds__(128) void flash_attn(const bf16_t* __restrict__ Q,
                                                  const bf16_t* __restrict__ K,
                                                  const bf16_t* __restrict__ Vt,
                                                  bf16_t* __restrict__ O) {
    __shared__ bf16_t Pbuf[4][16 * 32];
    const int waveId = threadIdx.x >> 5;
    const int lane   = threadIdx.x & 31;
    const int bh     = blockIdx.y;
    const int qRow   = blockIdx.x * 64 + waveId * 16;

    const bf16_t* Qh = Q  + (size_t)bh * SEQ * DK;
    const bf16_t* Kh = K  + (size_t)bh * SEQ * DK;
    const bf16_t* Vh = Vt + (size_t)bh * DK * SEQ;

    v16bf qf[4];
    #pragma unroll
    for (int c = 0; c < 4; ++c) qf[c] = load_fragA(Qh, qRow, DK, c * 32);

    v8f zero = {};
    v8f o[8];
    float mi[8], li[8];
    #pragma unroll
    for (int t = 0; t < 8; ++t) o[t] = zero;
    #pragma unroll
    for (int r = 0; r < 8; ++r) { mi[r] = -3.0e38f; li[r] = 0.f; }

    const int sel = lane >> 4, nl = lane & 15;
    bf16_t* pb = Pbuf[waveId];

    for (int kt = 0; kt < SEQ; kt += 32) {
        if (kt + 32 < SEQ)  // hint next K tile into cache (global_prefetch_b8)
            __builtin_prefetch(Kh + (size_t)(kt + 32) * DK + (lane << 3), 0, 1);

        v8f s0 = zero, s1 = zero;
        #pragma unroll
        for (int c = 0; c < 4; ++c) {
            v16bf b0 = load_fragB(Kh, kt,      DK, c * 32);
            v16bf b1 = load_fragB(Kh, kt + 16, DK, c * 32);
            s0 = wmma_bf16(qf[c], b0, s0);
            s1 = wmma_bf16(qf[c], b1, s1);
        }
        #pragma unroll
        for (int r = 0; r < 8; ++r) {
            float pm = fmaxf(s0[r], s1[r]);
            pm = fmaxf(pm, __shfl_xor(pm, 1));
            pm = fmaxf(pm, __shfl_xor(pm, 2));
            pm = fmaxf(pm, __shfl_xor(pm, 4));
            pm = fmaxf(pm, __shfl_xor(pm, 8));
            const float mn = fmaxf(mi[r], pm);
            const float scale = __expf(mi[r] - mn);
            const float p0 = __expf(s0[r] - mn);
            const float p1 = __expf(s1[r] - mn);
            float rs = p0 + p1;
            rs += __shfl_xor(rs, 1);
            rs += __shfl_xor(rs, 2);
            rs += __shfl_xor(rs, 4);
            rs += __shfl_xor(rs, 8);
            li[r] = li[r] * scale + rs;
            mi[r] = mn;
            #pragma unroll
            for (int t = 0; t < 8; ++t) o[t][r] *= scale;
            const int mloc = r + sel * 8;
            pb[mloc * 32 + nl]      = (bf16_t)p0;
            pb[mloc * 32 + 16 + nl] = (bf16_t)p1;
        }
        // C-layout -> A-layout through LDS (per-wave, DS ops in-order per wave)
        v16bf pf = load_fragA(pb, 0, 32, 0);
        #pragma unroll
        for (int t = 0; t < 8; ++t) {
            v16bf bv = load_fragB(Vh, t * 16, SEQ, kt);
            o[t] = wmma_bf16(pf, bv, o[t]);
        }
    }

    const int bb = bh >> 4, h = bh & 15;
    #pragma unroll
    for (int r = 0; r < 8; ++r) {
        const float inv = 1.0f / li[r];
        const int s = qRow + r + sel * 8;
        const size_t rowBase = ((size_t)bb * SEQ + s) * D_MODEL + (size_t)h * DK;
        #pragma unroll
        for (int t = 0; t < 8; ++t)
            O[rowBase + t * 16 + nl] = (bf16_t)(o[t][r] * inv);
    }
}

// ---------------------------------------------------------------- launch
extern "C" void kernel_launch(void* const* d_in, const int* in_sizes, int n_in,
                              void* d_out, int out_size, void* d_ws, size_t ws_size,
                              hipStream_t stream) {
    (void)in_sizes; (void)n_in; (void)out_size; (void)ws_size;
    const float* x  = (const float*)d_in[0];
    const float* wq = (const float*)d_in[1];
    const float* wk = (const float*)d_in[2];
    const float* wv = (const float*)d_in[3];
    const float* wo = (const float*)d_in[4];

    char* ws = (char*)d_ws;
    const size_t XB = (size_t)MTOT * D_MODEL;       // 16,777,216 elems
    const size_t WB = (size_t)D_MODEL * D_MODEL;    //  4,194,304 elems
    bf16_t* xb  = (bf16_t*)(ws);
    bf16_t* wqb = (bf16_t*)(ws + XB * 2);
    bf16_t* wkb = (bf16_t*)(ws + XB * 2 + WB * 2);
    bf16_t* wvb = (bf16_t*)(ws + XB * 2 + WB * 4);
    bf16_t* wob = (bf16_t*)(ws + XB * 2 + WB * 6);
    bf16_t* Qb  = (bf16_t*)(ws + XB * 2 + WB * 8);
    bf16_t* Kb  = (bf16_t*)(ws + XB * 4 + WB * 8);
    bf16_t* Vtb = (bf16_t*)(ws + XB * 6 + WB * 8);
    bf16_t* Ob  = (bf16_t*)(ws + XB * 8 + WB * 8);  // ends at ~192 MB

    cvt_f32_bf16<<<dim3(XB / 1024), 256, 0, stream>>>(x,  xb,  XB);
    cvt_f32_bf16<<<dim3(WB / 1024), 256, 0, stream>>>(wq, wqb, WB);
    cvt_f32_bf16<<<dim3(WB / 1024), 256, 0, stream>>>(wk, wkb, WB);
    cvt_f32_bf16<<<dim3(WB / 1024), 256, 0, stream>>>(wv, wvb, WB);
    cvt_f32_bf16<<<dim3(WB / 1024), 256, 0, stream>>>(wo, wob, WB);

    dim3 gg(MTOT / 128, D_MODEL / 128);
    gemm_bf16<0><<<gg, 128, 0, stream>>>(xb, wqb, Qb);
    gemm_bf16<1><<<gg, 128, 0, stream>>>(xb, wkb, Kb);
    gemm_bf16<2><<<gg, 128, 0, stream>>>(xb, wvb, Vtb);

    flash_attn<<<dim3(SEQ / 64, BATCH * HEADS), 128, 0, stream>>>(Qb, Kb, Vtb, Ob);

    gemm_bf16<3><<<gg, 128, 0, stream>>>(Ob, wob, d_out);
}
